// GraphConvnet_48069273977470
// MI455X (gfx1250) — matrix operand order repaired and checked by reading, not compiled, not run on hardware
//
#include <hip/hip_runtime.h>
#include <cstdint>
#include <cstddef>

// ---------------------------------------------------------------------------
// GraphMultisetTransformer GNN forward for MI455X (gfx1250, wave32, WMMA).
// fp32 storage in HBM; f16 conversion at the WMMA boundary; f32 accumulate.
// A-tiles staged into LDS by the Tensor Data Mover (TDM) when available.
// ---------------------------------------------------------------------------

typedef __attribute__((ext_vector_type(16))) _Float16 v16h;
typedef __attribute__((ext_vector_type(2)))  _Float16 h2v;
typedef __attribute__((ext_vector_type(8)))  float    v8f;
typedef __attribute__((ext_vector_type(4)))  unsigned int u32x4;
typedef __attribute__((ext_vector_type(8)))  int      i32x8;
typedef __attribute__((ext_vector_type(4)))  int      i32x4;

#if __has_builtin(__builtin_amdgcn_tensor_load_to_lds)
#define GEMM_USE_TDM 1
#endif

static constexpr int    kN     = 131072;     // nodes
static constexpr int    kE     = 1048576;    // directed edges
static constexpr int    kB     = 512;        // graphs
static constexpr int    kNPG   = 256;        // nodes per graph
static constexpr int    kNQ    = 75;         // seeds in PMA1 / rows in SAB
static constexpr float  kScale = 0.08838834764831845f; // 1/sqrt(128)

// ===========================================================================
// Generic GEMM: C[M x Nc] = A[M x K] @ W[K x Nc] (+ bias), fp32 in/out,
// f16 WMMA compute. Block = 256 threads = 8 waves. Block tile: 128 x 32.
// Each wave owns a 16x32 C tile (two 16x16 WMMA accumulators sharing A).
// grid = (ceil(M/128), Nc/32). K multiple of 32, Nc multiple of 32.
// ===========================================================================
__global__ __launch_bounds__(256) void gemm_wmma_f16(
    const float* __restrict__ A, const float* __restrict__ W,
    const float* __restrict__ bias, float* __restrict__ C,
    int M, int K, int Nc)
{
  __shared__ float    Af[128 * 33];   // f32 A tile, stride 33 (TDM pad = 1 dword / 32)
  __shared__ _Float16 Bh[32 * 34];    // f16 B tile, transposed [col][k], stride 34

  const int tid  = threadIdx.x;
  const int w    = tid >> 5;        // wave id 0..7
  const int lane = tid & 31;
  const int hi   = lane >> 4;       // 0: lanes 0-15, 1: lanes 16-31
  const int lr   = lane & 15;
  const int row0 = blockIdx.x * 128;
  const int col0 = blockIdx.y * 32;

  v8f acc0 = {}, acc1 = {};

  for (int k0 = 0; k0 < K; k0 += 32) {
#if defined(GEMM_USE_TDM)
    // ------- A tile via Tensor Data Mover: global[128x32 f32] -> LDS -------
    // D# (ISA 08_async_tensor 8.3/8.4): 2D tile, data_size=4B,
    // tile_dim0=32 cols, tile_dim1=128 rows, tensor dims = remaining extent
    // (hardware zero-fills OOB rows), pad 1 dword every 32 dwords -> stride 33.
    if (w == 0) {
      unsigned long long ga =
          (unsigned long long)(const void*)(A + (size_t)row0 * K + k0);
      unsigned ldsoff = (unsigned)(size_t)(void*)Af;  // low 32 bits = LDS offset
      unsigned dim0 = (unsigned)(K - k0);             // remaining cols
      unsigned dim1 = (unsigned)(M - row0);           // remaining rows
      u32x4 g0;
      g0[0] = 1u;                                     // count=1, user mode
      g0[1] = ldsoff;                                 // lds_addr
      g0[2] = (unsigned)(ga & 0xffffffffu);           // global_addr[31:0]
      g0[3] = ((unsigned)(ga >> 32) & 0x01ffffffu) | (2u << 30); // addr[56:32]|type=2
      i32x8 g1;
      g1[0] = (int)((2u << 16) | (1u << 20) | (4u << 22)); // 4B, pad_en, every 32 dw, +1 dw
      g1[1] = (int)((dim0 & 0xffffu) << 16);               // tensor_dim0 lo16
      g1[2] = (int)(((dim0 >> 16) & 0xffffu) | ((dim1 & 0xffffu) << 16));
      g1[3] = (int)(((dim1 >> 16) & 0xffffu) | (32u << 16)); // dim1 hi | tile_dim0=32
      g1[4] = 128;                                          // tile_dim1=128, tile_dim2=0
      g1[5] = (int)K;                                       // tensor_dim0_stride lo32
      g1[6] = 0;                                            // stride hi | dim1_stride lo
      g1[7] = 0;
      i32x4 gz4 = {0, 0, 0, 0};
      i32x8 gz8 = {0, 0, 0, 0, 0, 0, 0, 0};
      // 6-arg form (clang-23 / therock-10.0 headers).
      __builtin_amdgcn_tensor_load_to_lds(g0, g1, gz4, gz4, gz8, 0);
    }
#else
    // ------- A tile fallback: wide vector loads, branch-free full tiles ----
    if (row0 + 128 <= M) {
      #pragma unroll
      for (int i = 0; i < 4; ++i) {
        int e = (i << 8) + tid;               // 1024 float4 loads
        int r = e >> 3, c4 = (e & 7) << 2;
        float4 v = *(const float4*)(A + (size_t)(row0 + r) * K + k0 + c4);
        float* p = &Af[r * 33 + c4];
        p[0] = v.x; p[1] = v.y; p[2] = v.z; p[3] = v.w;
      }
    } else {
      for (int e = tid; e < 128 * 32; e += 256) {
        int r = e >> 5, c = e & 31;
        int gr = row0 + r;
        Af[r * 33 + c] = (gr < M) ? A[(size_t)gr * K + k0 + c] : 0.0f;
      }
    }
#endif
    // ------- B tile: 32(K) x 32(cols) f32 -> f16 transposed LDS ------------
    #pragma unroll
    for (int i = 0; i < 4; ++i) {
      int e = (i << 8) + tid;
      int r = e >> 5, c = e & 31;           // r = k index, c = column
      Bh[c * 34 + r] = (_Float16)W[(size_t)(k0 + r) * Nc + col0 + c];
    }
#if defined(GEMM_USE_TDM)
    if (w == 0) __builtin_amdgcn_s_wait_tensorcnt(0);
#endif
    __syncthreads();

    // ------- pack operands in contiguous k order (same for A and B) ---------------------
    // A 16x32: lane row = lane&15; K(j) = j + 16*(lane>=16)
    // B 32x16: lane col = lane&15; K(j) = j + 16*(lane>=16)
    v16h va, vb0, vb1;
    const int arow = (w << 4) + lr;
#pragma unroll
    for (int j = 0; j < 16; ++j) {
      int ka = j + (hi << 4);
      va[j] = (_Float16)Af[arow * 33 + ka];
    }
    const h2v* b0 = (const h2v*)&Bh[lr * 34];
    const h2v* b1 = (const h2v*)&Bh[(lr + 16) * 34];
#pragma unroll
    for (int p = 0; p < 8; ++p) {
      int kb2 = p + (hi << 3);
      h2v x0 = b0[kb2], x1 = b1[kb2];
      vb0[2 * p] = x0[0]; vb0[2 * p + 1] = x0[1];
      vb1[2 * p] = x1[0]; vb1[2 * p + 1] = x1[1];
    }
    // EXEC all-ones here: no divergence surrounds the WMMAs.
    acc0 = __builtin_amdgcn_wmma_f32_16x16x32_f16(
        false, va, false, vb0, (short)0, acc0, false, false);
    acc1 = __builtin_amdgcn_wmma_f32_16x16x32_f16(
        false, va, false, vb1, (short)0, acc1, false, false);
    __syncthreads();
  }

  const float bc0 = bias ? bias[col0 + lr] : 0.0f;
  const float bc1 = bias ? bias[col0 + 16 + lr] : 0.0f;
  // C/D layout: VGPR r -> row r + 8*(lane>=16); col = lane&15.
#pragma unroll
  for (int r = 0; r < 8; ++r) {
    int gr = row0 + (w << 4) + r + (hi << 3);
    if (gr < M) {
      C[(size_t)gr * Nc + col0 + lr]      = acc0[r] + bc0;
      C[(size_t)gr * Nc + col0 + 16 + lr] = acc1[r] + bc1;
    }
  }
}

// ===========================================================================
// Small helpers
// ===========================================================================
__global__ void fill_f(float* p, float v, size_t n) {
  size_t i = (size_t)blockIdx.x * blockDim.x + threadIdx.x;
  if (i < n) p[i] = v;
}

__global__ void deg_edge(const int* __restrict__ dst, float* __restrict__ deg) {
  int e = blockIdx.x * blockDim.x + threadIdx.x;
  if (e < kE) atomicAdd(&deg[dst[e]], 1.0f);
}

__global__ void deg_finalize(float* d) {   // dis = rsqrt(deg); deg >= 1 (self loops)
  int i = blockIdx.x * blockDim.x + threadIdx.x;
  if (i < kN) d[i] = rsqrtf(d[i]);
}

// Embedding: h[i,c] = b[c] + sum_f x[i,f] * W[f,c].  grid=kN, block=128.
__global__ __launch_bounds__(128) void emb_kernel(
    const float* __restrict__ x, const float* __restrict__ W,
    const float* __restrict__ b, float* __restrict__ h)
{
  __shared__ float xs[42];
  const int i = blockIdx.x, c = threadIdx.x;
  if (c < 42) xs[c] = x[(size_t)i * 42 + c];
  __syncthreads();
  float acc = b[c];
#pragma unroll
  for (int f = 0; f < 42; ++f) acc += xs[f] * W[f * 128 + c];
  h[(size_t)i * 128 + c] = acc;
}

// Column mean/var accumulation (128 cols). block=128, each block sums 64 rows.
__global__ __launch_bounds__(128) void col_accum(
    const float* __restrict__ h, float* __restrict__ sum, float* __restrict__ ssq)
{
  const int c = threadIdx.x;
  const int r0 = blockIdx.x * 64;
  float s = 0.f, q = 0.f;
  for (int r = 0; r < 64; ++r) {
    float v = h[(size_t)(r0 + r) * 128 + c];
    s += v; q += v * v;
  }
  atomicAdd(&sum[c], s);
  atomicAdd(&ssq[c], q);
}

__global__ void col_norm(float* __restrict__ h, const float* __restrict__ sum,
                         const float* __restrict__ ssq, float invN, size_t total)
{
  size_t i = (size_t)blockIdx.x * blockDim.x + threadIdx.x;
  if (i >= total) return;
  int c = (int)(i & 127);
  float mu = sum[c] * invN;
  float var = fmaxf(ssq[c] * invN - mu * mu, 0.0f);
  h[i] = (h[i] - mu) / (sqrtf(var) + 1e-8f);
}

// GCN scatter init: out = bias + tmp * dis^2 (self-loop term), non-atomic.
__global__ void gcn_self(const float* __restrict__ tmp, const float* __restrict__ dis,
                         const float* __restrict__ bias, float* __restrict__ out,
                         int fo_log2, size_t total)
{
  size_t i = (size_t)blockIdx.x * blockDim.x + threadIdx.x;
  if (i >= total) return;
  int node = (int)(i >> fo_log2);
  int c = (int)(i & ((1u << fo_log2) - 1));
  float ds = dis[node];
  out[i] = bias[c] + tmp[i] * ds * ds;
}

// GCN edge scatter: out[dst] += tmp[src] * dis[src]*dis[dst], 4 cols/thread.
__global__ void gcn_edge(const int* __restrict__ src, const int* __restrict__ dst,
                         const float* __restrict__ tmp, const float* __restrict__ dis,
                         float* __restrict__ out, int fo_log2)
{
  const int csh = fo_log2 - 2;                 // log2(fo/4)
  long long idx = (long long)blockIdx.x * blockDim.x + threadIdx.x;
  long long e = idx >> csh;
  if (e >= kE) return;
  int c4 = (int)((idx & ((1ll << csh) - 1)) << 2);
  int s = src[e], d = dst[e];
  float coef = dis[s] * dis[d];
  const float4 v = *(const float4*)(tmp + ((size_t)s << fo_log2) + c4);
  float* o = out + ((size_t)d << fo_log2) + c4;
  atomicAdd(o + 0, v.x * coef);
  atomicAdd(o + 1, v.y * coef);
  atomicAdd(o + 2, v.z * coef);
  atomicAdd(o + 3, v.w * coef);
}

__global__ void leaky_inplace(float* p, size_t n) {
  size_t i = (size_t)blockIdx.x * blockDim.x + threadIdx.x;
  if (i >= n) return;
  float v = p[i];
  p[i] = v > 0.0f ? v : 0.01f * v;
}

// h = o + relu(t)
__global__ void add_relu(float* __restrict__ out, const float* __restrict__ o,
                         const float* __restrict__ t, size_t n)
{
  size_t i = (size_t)blockIdx.x * blockDim.x + threadIdx.x;
  if (i >= n) return;
  out[i] = o[i] + fmaxf(t[i], 0.0f);
}

// ===========================================================================
// Fused MAB attention core. Softmax is over the QUERY axis (faithful to PyG
// GMT), i.e. independent per (b,h,k) -> stream keys, no score tensor.
// One block per (b,h); O[b,q,h*16+d] = Q(q) + sum_k softmax_q(s)_qk * V[b,k].
// q_bstride = 0 for a shared (seed) Q, else nq*128.
// ===========================================================================
__global__ __launch_bounds__(256) void att_fused(
    const float* __restrict__ Qf, int q_bstride,
    const float* __restrict__ Kf, const float* __restrict__ Vf,
    float* __restrict__ O, int nk)
{
  __shared__ float qs[kNQ * 16];
  __shared__ float oacc[kNQ * 16];
  const int b = blockIdx.x, h = blockIdx.y, tid = threadIdx.x;
  const float* Qb = Qf + (size_t)b * q_bstride;

  for (int i = tid; i < kNQ * 16; i += 256) {
    int q = i >> 4, d = i & 15;
    float qv = Qb[q * 128 + h * 16 + d];
    qs[i] = qv;
    oacc[i] = qv;                      // o = qh + sum_k A*vh
  }
  __syncthreads();

  for (int k = tid; k < nk; k += 256) {
    const float* kp = Kf + ((size_t)b * nk + k) * 128 + h * 16;
    const float* vp = Vf + ((size_t)b * nk + k) * 128 + h * 16;
    float kr[16], vr[16];
#pragma unroll
    for (int d = 0; d < 16; ++d) { kr[d] = kp[d]; vr[d] = vp[d]; }

    float mx = -3.0e38f;
#pragma unroll 1
    for (int q = 0; q < kNQ; ++q) {
      float s = 0.f;
#pragma unroll
      for (int d = 0; d < 16; ++d) s += qs[q * 16 + d] * kr[d];
      mx = fmaxf(mx, s * kScale);
    }
    float denom = 0.f;
#pragma unroll 1
    for (int q = 0; q < kNQ; ++q) {
      float s = 0.f;
#pragma unroll
      for (int d = 0; d < 16; ++d) s += qs[q * 16 + d] * kr[d];
      denom += __expf(s * kScale - mx);
    }
    float inv = 1.0f / denom;
#pragma unroll 1
    for (int q = 0; q < kNQ; ++q) {
      float s = 0.f;
#pragma unroll
      for (int d = 0; d < 16; ++d) s += qs[q * 16 + d] * kr[d];
      float a = __expf(s * kScale - mx) * inv;
#pragma unroll
      for (int d = 0; d < 16; ++d) atomicAdd(&oacc[q * 16 + d], a * vr[d]);
    }
  }
  __syncthreads();

  for (int i = tid; i < kNQ * 16; i += 256) {
    int q = i >> 4, d = i & 15;
    O[((size_t)b * kNQ + q) * 128 + h * 16 + d] = oacc[i];
  }
}

// PMA2: softmax over size-1 query dim => A == 1 => o = q3 + sum_k v3[b,k].
__global__ __launch_bounds__(128) void pma2_sum(
    const float* __restrict__ q3, const float* __restrict__ v3, float* __restrict__ o3)
{
  const int b = blockIdx.x, c = threadIdx.x;
  float acc = q3[c];
  const float* vp = v3 + (size_t)b * kNQ * 128 + c;
#pragma unroll 1
  for (int k = 0; k < kNQ; ++k) acc += vp[k * 128];
  o3[(size_t)b * 128 + c] = acc;
}

// ===========================================================================
// Host-side orchestration
// ===========================================================================
// Input order: depth-first insertion-order flatten of setup_inputs():
// x, edge_index, batch, emb(W,b), convs[3](W,b), lin1(W,b),
// pma1{q,k,v,o}(W,b), S1, sab{q,k,v,o}(W,b), pma2{q,k,v,o}(W,b), S2,
// lin2(W,b), flat(W,b)  -> 43 leaves.
enum {
  IN_X = 0, IN_EI = 1, IN_BATCH = 2,
  P_EMB_W = 3, P_EMB_B = 4,
  P_C1_W = 5, P_C1_B = 6, P_C2_W = 7, P_C2_B = 8, P_C3_W = 9, P_C3_B = 10,
  P_LIN1_W = 11, P_LIN1_B = 12,
  P_P1_QW = 13, P_P1_QB = 14, P_P1_KW = 15, P_P1_KB = 16,
  P_P1_VW = 17, P_P1_VB = 18, P_P1_OW = 19, P_P1_OB = 20,
  P_S1 = 21,
  P_SB_QW = 22, P_SB_QB = 23, P_SB_KW = 24, P_SB_KB = 25,
  P_SB_VW = 26, P_SB_VB = 27, P_SB_OW = 28, P_SB_OB = 29,
  P_P2_QW = 30, P_P2_QB = 31, P_P2_KW = 32, P_P2_KB = 33,
  P_P2_VW = 34, P_P2_VB = 35, P_P2_OW = 36, P_P2_OB = 37,
  P_S2 = 38, P_LIN2_W = 39, P_LIN2_B = 40, P_FLAT_W = 41, P_FLAT_B = 42
};

extern "C" void kernel_launch(void* const* d_in, const int* in_sizes, int n_in,
                              void* d_out, int out_size, void* d_ws, size_t ws_size,
                              hipStream_t stream)
{
  (void)in_sizes; (void)n_in; (void)out_size; (void)ws_size;
  auto F = [&](int i) { return (const float*)d_in[i]; };
  const float* x  = F(IN_X);
  const int* ei   = (const int*)d_in[IN_EI];
  const int* src  = ei;
  const int* dst  = ei + kE;

  char* ws = (char*)d_ws;
  // Workspace map (258 MB total, phase-wise reuse of the four 64MB buffers).
  float* dis  = (float*)(ws + 0x00000000);           // kN floats
  float* csum = (float*)(ws + 0x00080000);           // 128
  float* cssq = (float*)(ws + 0x00080200);           // 128
  float* qS   = (float*)(ws + 0x00081000);           // 75*128
  float* q3   = (float*)(ws + 0x00090000);           // 128
  float* o3   = (float*)(ws + 0x000A0000);           // 512*128
  float* t3   = (float*)(ws + 0x000E0000);           // 512*128
  float* h3   = (float*)(ws + 0x00120000);           // 512*128
  float* t4   = (float*)(ws + 0x00160000);           // 512*128
  const size_t BIGSZ = 64ull << 20;
  float* BIG0 = (float*)(ws + 0x00200000);
  float* BIG1 = (float*)((char*)BIG0 + BIGSZ);
  float* BIG2 = (float*)((char*)BIG1 + BIGSZ);
  float* BIG3 = (float*)((char*)BIG2 + BIGSZ);
  float* BIG0h = (float*)((char*)BIG0 + (32ull << 20));
  float* BIG3h = (float*)((char*)BIG3 + (32ull << 20));

  auto cdiv = [](size_t a, size_t b) { return (unsigned)((a + b - 1) / b); };
  auto gemm = [&](const float* A, const float* W, const float* bias, float* C,
                  int M, int K, int Nc) {
    dim3 g(cdiv(M, 128), Nc / 32);
    gemm_wmma_f16<<<g, 256, 0, stream>>>(A, W, bias, C, M, K, Nc);
  };
  auto gcn_scatter = [&](const float* tmp, const float* bias, float* out, int lg2) {
    size_t tot = (size_t)kN << lg2;
    gcn_self<<<cdiv(tot, 256), 256, 0, stream>>>(tmp, dis, bias, out, lg2, tot);
    size_t eth = (size_t)kE << (lg2 - 2);
    gcn_edge<<<cdiv(eth, 256), 256, 0, stream>>>(src, dst, tmp, dis, out, lg2);
  };
  auto leaky = [&](float* p, size_t n) {
    leaky_inplace<<<cdiv(n, 256), 256, 0, stream>>>(p, n);
  };

  // --- degree -> dis = rsqrt(deg) (self-loops give deg >= 1) ---------------
  fill_f<<<cdiv(kN, 256), 256, 0, stream>>>(dis, 1.0f, kN);
  deg_edge<<<cdiv(kE, 256), 256, 0, stream>>>(dst, dis);
  deg_finalize<<<cdiv(kN, 256), 256, 0, stream>>>(dis);

  // --- embedding + detached per-feature z-score ---------------------------
  emb_kernel<<<kN, 128, 0, stream>>>(x, F(P_EMB_W), F(P_EMB_B), BIG0);
  fill_f<<<1, 256, 0, stream>>>(csum, 0.0f, 256);    // csum+cssq contiguous
  col_accum<<<kN / 64, 128, 0, stream>>>(BIG0, csum, cssq);
  col_norm<<<cdiv((size_t)kN * 128, 256), 256, 0, stream>>>(
      BIG0, csum, cssq, 1.0f / kN, (size_t)kN * 128);

  // --- 3 GCN convs + leaky-relu -------------------------------------------
  gemm(BIG0, F(P_C1_W), nullptr, BIG1, kN, 128, 32);
  gcn_scatter(BIG1, F(P_C1_B), BIG2, 5);  leaky(BIG2, (size_t)kN * 32);
  gemm(BIG2, F(P_C2_W), nullptr, BIG1, kN, 32, 64);
  gcn_scatter(BIG1, F(P_C2_B), BIG0, 6);  leaky(BIG0, (size_t)kN * 64);
  gemm(BIG0, F(P_C3_W), nullptr, BIG1, kN, 64, 128);
  gcn_scatter(BIG1, F(P_C3_B), BIG2, 7);  leaky(BIG2, (size_t)kN * 128);

  // --- lin1, then K/V via GCN (no activation) -----------------------------
  gemm(BIG2, F(P_LIN1_W), F(P_LIN1_B), BIG3, kN, 128, 128);      // x1 = BIG3
  gemm(BIG3, F(P_P1_KW), nullptr, BIG1, kN, 128, 128);
  gcn_scatter(BIG1, F(P_P1_KB), BIG0, 7);                         // K = BIG0
  gemm(BIG3, F(P_P1_VW), nullptr, BIG1, kN, 128, 128);
  gcn_scatter(BIG1, F(P_P1_VB), BIG2, 7);                         // V = BIG2

  // --- PMA1: q = S1@Wq+b (shared across graphs), fused attention ----------
  gemm(F(P_S1), F(P_P1_QW), F(P_P1_QB), qS, kNQ, 128, 128);
  {
    dim3 g(kB, 8);
    att_fused<<<g, 256, 0, stream>>>(qS, 0, BIG0, BIG2, BIG1, kNPG); // O1=BIG1
  }
  gemm(BIG1, F(P_P1_OW), F(P_P1_OB), BIG3, kB * kNQ, 128, 128);   // T = BIG3
  add_relu<<<cdiv((size_t)kB * kNQ * 128, 256), 256, 0, stream>>>(
      BIG3h, BIG1, BIG3, (size_t)kB * kNQ * 128);                 // h1 = BIG3h

  // --- SAB: self-attention over h1 ----------------------------------------
  gemm(BIG3h, F(P_SB_QW), F(P_SB_QB), BIG0,  kB * kNQ, 128, 128); // q2
  gemm(BIG3h, F(P_SB_KW), F(P_SB_KB), BIG0h, kB * kNQ, 128, 128); // k2
  gemm(BIG3h, F(P_SB_VW), F(P_SB_VB), BIG1,  kB * kNQ, 128, 128); // v2
  {
    dim3 g(kB, 8);
    att_fused<<<g, 256, 0, stream>>>(BIG0, kNQ * 128, BIG0h, BIG1, BIG2, kNQ); // O2=BIG2
  }
  gemm(BIG2, F(P_SB_OW), F(P_SB_OB), BIG1, kB * kNQ, 128, 128);   // T = BIG1
  add_relu<<<cdiv((size_t)kB * kNQ * 128, 256), 256, 0, stream>>>(
      BIG0, BIG2, BIG1, (size_t)kB * kNQ * 128);                  // h2 = BIG0

  // --- PMA2: nq=1 => softmax over q is identity-to-1, K projection dead ---
  gemm(F(P_S2), F(P_P2_QW), F(P_P2_QB), q3, 1, 128, 128);
  gemm(BIG0, F(P_P2_VW), F(P_P2_VB), BIG1, kB * kNQ, 128, 128);   // v3 = BIG1
  pma2_sum<<<kB, 128, 0, stream>>>(q3, BIG1, o3);
  gemm(o3, F(P_P2_OW), F(P_P2_OB), t3, kB, 128, 128);
  add_relu<<<cdiv((size_t)kB * 128, 256), 256, 0, stream>>>(
      h3, o3, t3, (size_t)kB * 128);

  // --- lin2 -> flat -> d_out ----------------------------------------------
  gemm(h3, F(P_LIN2_W), F(P_LIN2_B), t4, kB, 128, 128);
  gemm(t4, F(P_FLAT_W), F(P_FLAT_B), (float*)d_out, kB, 128, 128);
}
